// LSTMClassifier_26757646254644
// MI455X (gfx1250) — matrix operand order repaired and checked by reading, not compiled, not run on hardware
//
#include <hip/hip_runtime.h>

// ---------------------------------------------------------------------------
// Fused 2-layer LSTM (H=64, T=256, B=4096, F=5) + FC head for gfx1250.
// One wave = 16 batch rows, state resident across all 256 timesteps.
// Gate GEMMs via v_wmma_f32_16x16x32_f16; weights pre-swizzled into LDS
// B-fragment layout; h->A-fragment transpose via LDS staging.
// ---------------------------------------------------------------------------

typedef __attribute__((ext_vector_type(16))) _Float16 v16h;
typedef __attribute__((ext_vector_type(8)))  _Float16 v8h;
typedef __attribute__((ext_vector_type(8)))  float    v8f;

#define T_STEPS 256
#define HID     64
#define GDIM    256
#define NF      5
#define BATCH   4096

// LDS layout (byte offsets), all 128B-aligned-compatible
#define OFF_WHH0 0        // 16 tiles * 2 ksteps * 32 lanes * 16 halfs = 32768 B
#define OFF_WIH0 32768    // 16 * 1 * 32 * 16 halfs                   = 16384 B
#define OFF_WIH1 49152    // 32768 B
#define OFF_WHH1 81920    // 32768 B
#define OFF_B0   114688   // 256 floats = 1024 B
#define OFF_B1   115712   // 1024 B
#define OFF_HST  116736   // 4 waves * 2 layers * 16*64 halfs = 16384 B
#define SMEM_BYTES 133120

__device__ __forceinline__ float tanh_fast(float x) {
#if __has_builtin(__builtin_amdgcn_tanhf)
  return __builtin_amdgcn_tanhf(x);          // v_tanh_f32 (gfx1250 trans op)
#else
  float e = __expf(2.0f * x);
  return 1.0f - 2.0f / (e + 1.0f);
#endif
}
__device__ __forceinline__ float sig_fast(float x) {
  return 0.5f * tanh_fast(0.5f * x) + 0.5f;
}

__device__ __forceinline__ v8f wmma16(v16h a, v16h b, v8f c) {
  // (neg_a, A, neg_b, B, c_mod, C, reuse_a, reuse_b)
  return __builtin_amdgcn_wmma_f32_16x16x32_f16(false, a, false, b, (short)0, c,
                                                false, false);
}

// B fragment: [fragIdx][lane][16 halfs], lane-contiguous 32 B
__device__ __forceinline__ v16h load_bfrag(const _Float16* base, int fragIdx,
                                           int lane) {
  return *(const v16h*)(base + ((size_t)fragIdx * 32 + lane) * 16);
}

// A fragment from row-major f16 staging [16 rows][64 K]:
// ISA 16-bit A 16x32 layout: lanes 0-15 hold K=base+0..7 & base+16..23,
// lanes 16-31 hold K=base+8..15 & base+24..31  (two contiguous 16B runs).
__device__ __forceinline__ v16h load_afrag(const _Float16* hst, int M, int g,
                                           int s) {
  const v8h lo = *(const v8h*)(hst + M * 64 + 32 * s + 8 * g);
  const v8h hi = *(const v8h*)(hst + M * 64 + 32 * s + 16 + 8 * g);
  v16h a;
#pragma unroll
  for (int j = 0; j < 8; ++j) { a[j] = lo[j]; a[j + 8] = hi[j]; }
  return a;
}

#define LDS_FENCE() asm volatile("s_wait_dscnt 0x0" ::: "memory")

__global__ __launch_bounds__(128, 1) void lstm_fused_kernel(
    const float* __restrict__ X,
    const float* __restrict__ Wih0, const float* __restrict__ Whh0,
    const float* __restrict__ bih0, const float* __restrict__ bhh0,
    const float* __restrict__ Wih1, const float* __restrict__ Whh1,
    const float* __restrict__ bih1, const float* __restrict__ bhh1,
    const float* __restrict__ Wfc,  const float* __restrict__ bfc,
    float* __restrict__ Out) {
  extern __shared__ __align__(128) char smem[];
  _Float16* whh0f = (_Float16*)(smem + OFF_WHH0);
  _Float16* wih0f = (_Float16*)(smem + OFF_WIH0);
  _Float16* wih1f = (_Float16*)(smem + OFF_WIH1);
  _Float16* whh1f = (_Float16*)(smem + OFF_WHH1);
  float*    b0    = (float*)   (smem + OFF_B0);
  float*    b1    = (float*)   (smem + OFF_B1);
  _Float16* hsta  = (_Float16*)(smem + OFF_HST);

  const int tid = threadIdx.x;

  // ---- prologue: swizzle weights into WMMA B-fragment layout in LDS ----
  // B[K][N] = W[N][K] (gates = h @ W^T). frag e: j=e[3:0], lane=e[8:4],
// s=e[9], n=e[13:10]; K = 32*s + 8*(lane>>4) + (j&7) + 16*(j>>3), N = 16*n + (lane&15).
  for (int e = tid; e < 16 * 2 * 32 * 16; e += 128) {
    int j = e & 15, l = (e >> 4) & 31, s = (e >> 9) & 1, n = e >> 10;
    int K = 32 * s + 8 * (l >> 4) + (j & 7) + 16 * (j >> 3);
    int N = 16 * n + (l & 15);
    whh0f[e] = (_Float16)Whh0[N * HID + K];
    whh1f[e] = (_Float16)Whh1[N * HID + K];
    wih1f[e] = (_Float16)Wih1[N * HID + K];
  }
  // layer-0 input projection: K padded 5 -> 32 with zeros
  for (int e = tid; e < 16 * 32 * 16; e += 128) {
    int j = e & 15, l = (e >> 4) & 31, n = e >> 9;
    int K = 16 * (l >> 4) + j;
    int N = 16 * n + (l & 15);
    wih0f[e] = (K < NF) ? (_Float16)Wih0[N * NF + K] : (_Float16)0.0f;
  }
  for (int q = tid; q < GDIM; q += 128) {
    b0[q] = bih0[q] + bhh0[q];
    b1[q] = bih1[q] + bhh1[q];
  }
  for (int e = tid; e < 4 * 2 * 16 * 64; e += 128) hsta[e] = (_Float16)0.0f;
  __syncthreads();

  const int lane = tid & 31;
  const int wv   = tid >> 5;
  const int g    = lane >> 4;   // half-wave group
  const int nl   = lane & 15;   // N / M index within tile
  const int rowbase = blockIdx.x * 64 + wv * 16;
  _Float16* h0st = hsta + (wv * 2 + 0) * (16 * 64);
  _Float16* h1st = hsta + (wv * 2 + 1) * (16 * 64);

  v8f c0[4], c1[4];
#pragma unroll
  for (int n = 0; n < 4; ++n)
#pragma unroll
    for (int r = 0; r < 8; ++r) { c0[n][r] = 0.0f; c1[n][r] = 0.0f; }

#pragma unroll 1
  for (int t = 0; t < T_STEPS; ++t) {
    // x A-fragment: only lanes 0-15 carry K=0..4 (rest of padded K is zero)
    v16h ax;
#pragma unroll
    for (int j = 0; j < 16; ++j) ax[j] = (_Float16)0.0f;
    if (g == 0) {
      const float* xp = X + ((size_t)(rowbase + nl) * T_STEPS + t) * NF;
#pragma unroll
      for (int j = 0; j < NF; ++j) ax[j] = (_Float16)xp[j];
    }

    LDS_FENCE();
    v16h ah0[2], ah1[2];
#pragma unroll
    for (int s = 0; s < 2; ++s) {
      ah0[s] = load_afrag(h0st, nl, g, s);
      ah1[s] = load_afrag(h1st, nl, g, s);
    }

    // -------- layer 0: gates = b0 + x@Wih0^T + h0@Whh0^T --------
#pragma unroll
    for (int n = 0; n < 4; ++n) {
      v8f acc[4];
#pragma unroll
      for (int q = 0; q < 4; ++q) {   // q: i,f,g,o
        int tn = 4 * q + n;
        float bv = b0[16 * tn + nl];
#pragma unroll
        for (int r = 0; r < 8; ++r) acc[q][r] = bv;
        acc[q] = wmma16(ax,     load_bfrag(wih0f, tn,         lane), acc[q]);
        acc[q] = wmma16(ah0[0], load_bfrag(whh0f, tn * 2 + 0, lane), acc[q]);
        acc[q] = wmma16(ah0[1], load_bfrag(whh0f, tn * 2 + 1, lane), acc[q]);
      }
#pragma unroll
      for (int r = 0; r < 8; ++r) {
        float iv = sig_fast(acc[0][r]);
        float fv = sig_fast(acc[1][r]);
        float gv = tanh_fast(acc[2][r]);
        float ov = sig_fast(acc[3][r]);
        float cv = fv * c0[n][r] + iv * gv;
        c0[n][r] = cv;
        float hv = ov * tanh_fast(cv);
        h0st[(r + 8 * g) * 64 + 16 * n + nl] = (_Float16)hv;
      }
    }

    LDS_FENCE();
    v16h ah0n[2];
#pragma unroll
    for (int s = 0; s < 2; ++s) ah0n[s] = load_afrag(h0st, nl, g, s);

    // -------- layer 1: gates = b1 + h0_t@Wih1^T + h1@Whh1^T --------
#pragma unroll
    for (int n = 0; n < 4; ++n) {
      v8f acc[4];
#pragma unroll
      for (int q = 0; q < 4; ++q) {
        int tn = 4 * q + n;
        float bv = b1[16 * tn + nl];
#pragma unroll
        for (int r = 0; r < 8; ++r) acc[q][r] = bv;
        acc[q] = wmma16(ah0n[0], load_bfrag(wih1f, tn * 2 + 0, lane), acc[q]);
        acc[q] = wmma16(ah0n[1], load_bfrag(wih1f, tn * 2 + 1, lane), acc[q]);
        acc[q] = wmma16(ah1[0],  load_bfrag(whh1f, tn * 2 + 0, lane), acc[q]);
        acc[q] = wmma16(ah1[1],  load_bfrag(whh1f, tn * 2 + 1, lane), acc[q]);
      }
#pragma unroll
      for (int r = 0; r < 8; ++r) {
        float iv = sig_fast(acc[0][r]);
        float fv = sig_fast(acc[1][r]);
        float gv = tanh_fast(acc[2][r]);
        float ov = sig_fast(acc[3][r]);
        float cv = fv * c1[n][r] + iv * gv;
        c1[n][r] = cv;
        float hv = ov * tanh_fast(cv);
        h1st[(r + 8 * g) * 64 + 16 * n + nl] = (_Float16)hv;
      }
    }
  }

  // -------- FC head: out[row] = h1_T @ Wfc^T + bfc  (16 rows/wave) --------
  LDS_FENCE();
  if (g == 0) {
    int row = nl;
    float a0 = bfc[0], a1 = bfc[1];
#pragma unroll 1
    for (int k = 0; k < HID; ++k) {
      float hv = (float)h1st[row * HID + k];
      a0 += hv * Wfc[k];
      a1 += hv * Wfc[HID + k];
    }
    size_t orow = (size_t)(rowbase + row) * 2;
    Out[orow + 0] = a0;
    Out[orow + 1] = a1;
  }
}

extern "C" void kernel_launch(void* const* d_in, const int* in_sizes, int n_in,
                              void* d_out, int out_size, void* d_ws,
                              size_t ws_size, hipStream_t stream) {
  (void)in_sizes; (void)n_in; (void)out_size; (void)d_ws; (void)ws_size;
  const float* X    = (const float*)d_in[0];
  const float* Wih0 = (const float*)d_in[1];
  const float* Whh0 = (const float*)d_in[2];
  const float* bih0 = (const float*)d_in[3];
  const float* bhh0 = (const float*)d_in[4];
  const float* Wih1 = (const float*)d_in[5];
  const float* Whh1 = (const float*)d_in[6];
  const float* bih1 = (const float*)d_in[7];
  const float* bhh1 = (const float*)d_in[8];
  const float* Wfc  = (const float*)d_in[9];
  const float* bfc  = (const float*)d_in[10];

  // allow >64KB dynamic LDS (WGP has 320KB)
  static bool attr_set = false;
  if (!attr_set) {
    (void)hipFuncSetAttribute((const void*)lstm_fused_kernel,
                              hipFuncAttributeMaxDynamicSharedMemorySize,
                              SMEM_BYTES);
    attr_set = true;
  }

  dim3 grid(BATCH / 64);   // 4 waves/block * 16 rows/wave
  dim3 block(128);
  lstm_fused_kernel<<<grid, block, SMEM_BYTES, stream>>>(
      X, Wih0, Whh0, bih0, bhh0, Wih1, Whh1, bih1, bhh1, Wfc, bfc,
      (float*)d_out);
}
